// ClBlockLoss_85383949844735
// MI455X (gfx1250) — matrix operand order repaired and checked by reading, not compiled, not run on hardware
//
#include <hip/hip_runtime.h>
#include <hip/hip_bf16.h>
#include <math.h>

typedef __attribute__((ext_vector_type(16))) _Float16 v16h;
typedef __attribute__((ext_vector_type(8)))  _Float16 v8h;
typedef __attribute__((ext_vector_type(4)))  _Float16 v4h;
typedef __attribute__((ext_vector_type(8)))  float    v8f;

#define N_T   4
#define BSZ   2048
#define DIM   128
#define KTOT  (N_T * BSZ)            // 8192 rows
#define KPOS  (N_T * (N_T - 1) * BSZ) // 24576
#define SQRT_INV_TEMP 1.41421356237f // sqrt(1/TEMP), TEMP=0.5
#define EPSV  1e-8f
#define NEG_BIG (-1e30f)

// ---------------------------------------------------------------------------
// Kernel 0: zero the scalar output (harness poisons d_out; atomics accumulate)
// ---------------------------------------------------------------------------
__global__ void ClBlockLoss_zero_kernel(float* out) { out[0] = 0.0f; }

// ---------------------------------------------------------------------------
// Kernel 1: row-normalize z (8192 x 128 f32) -> snh (f16, scaled by sqrt(2))
// One wave per row; full-wave butterfly reduction of sum-of-squares.
// ---------------------------------------------------------------------------
__global__ __launch_bounds__(256) void ClBlockLoss_normalize_kernel(
    const float* __restrict__ z, _Float16* __restrict__ snh) {
  const int lane = threadIdx.x & 31;
  const int row  = blockIdx.x * 8 + (threadIdx.x >> 5);
  const float4 v = reinterpret_cast<const float4*>(z + (size_t)row * DIM)[lane];
  float ss = v.x * v.x + v.y * v.y + v.z * v.z + v.w * v.w;
#pragma unroll
  for (int off = 1; off < 32; off <<= 1) ss += __shfl_xor(ss, off, 32);
  const float scale = SQRT_INV_TEMP / fmaxf(sqrtf(ss), EPSV);
  v4h h;
  h.x = (_Float16)(v.x * scale);
  h.y = (_Float16)(v.y * scale);
  h.z = (_Float16)(v.z * scale);
  h.w = (_Float16)(v.w * scale);
  reinterpret_cast<v4h*>(snh + (size_t)row * DIM)[lane] = h;
}

// ---------------------------------------------------------------------------
// Kernel 2: fused sim-matrix WMMA + masked streaming logsumexp + loss.
// One wave per 16-row strip; A fragments resident in VGPRs; streams all
// 512 column tiles (B) out of L2; online (m,S) per lane per row, butterfly
// merge across the 16 column residues; one atomicAdd per wave.
// ---------------------------------------------------------------------------
__global__ __launch_bounds__(256) void ClBlockLoss_simloss_kernel(
    const _Float16* __restrict__ snh, float* __restrict__ out) {
  const int lane  = threadIdx.x & 31;
  const int wave  = threadIdx.x >> 5;
  const int strip = blockIdx.x * 8 + wave;   // 0..511
  const int r0    = strip * 16;
  const int half  = lane >> 4;               // 0: lanes 0-15, 1: lanes 16-31
  const int l16   = lane & 15;

  // --- Load A fragments for all 4 k-steps (16-bit A 16x32, this code's k order, same as B):
  // lane holds row M=l16; halves 0/1 hold K 0-15 / 16-31 (contiguous).
  v16h a[4];
  {
    const _Float16* arow = snh + (size_t)(r0 + l16) * DIM;
#pragma unroll
    for (int t = 0; t < 4; ++t) {
      const int kk = t * 32;
      v8h lo = *reinterpret_cast<const v8h*>(arow + kk + half * 16);
      v8h hi = *reinterpret_cast<const v8h*>(arow + kk + half * 16 + 8);
      a[t] = __builtin_shufflevector(lo, hi, 0, 1, 2, 3, 4, 5, 6, 7,
                                     8, 9, 10, 11, 12, 13, 14, 15);
    }
  }

  // Per-lane streaming LSE state for the 8 rows this lane's C-VGPRs cover.
  float m[8], S[8];
#pragma unroll
  for (int g = 0; g < 8; ++g) { m[g] = NEG_BIG; S[g] = 0.0f; }
  float p0 = 0.0f, p1 = 0.0f, p2 = 0.0f, p3 = 0.0f;  // pos_all[q], q=0..3

  // Diagonal element (M==N) for row M=l16 lives at VGPR g_diag if valid.
  const int g_diag = l16 - 8 * half;              // valid iff in [0,8)
  const int rmod   = r0 & (BSZ - 1);

  for (int c0 = 0; c0 < KTOT; c0 += 16) {
    // B fragment (32x16, K x N): lane = column N -> row snh[c0+N];
    // halves hold K 0-15 / 16-31 as 16 contiguous f16 (32B aligned).
    const _Float16* brow = snh + (size_t)(c0 + l16) * DIM + half * 16;
    if (c0 + 16 < KTOT)
      __builtin_prefetch(snh + (size_t)(c0 + 16 + l16) * DIM, 0, 3);

    v8f c = {};
#pragma unroll
    for (int t = 0; t < 4; ++t) {
      v16h b = *reinterpret_cast<const v16h*>(brow + t * 32);
      c = __builtin_amdgcn_wmma_f32_16x16x32_f16(
          /*neg_a=*/false, a[t], /*neg_b=*/false, b,
          /*c_mod=*/(short)0, c, /*reuse_a=*/false, /*reuse_b=*/false);
    }

    const bool diag = ((c0 ^ r0) & (BSZ - 1)) == 0;  // same r-residue tile
    const int  q    = c0 >> 11;                      // column view index

#pragma unroll
    for (int g = 0; g < 8; ++g) {
      float xv = c[g];
      const bool msk = diag && (g == g_diag);        // pos_all / self element
      p0 = (msk && q == 0) ? xv : p0;
      p1 = (msk && q == 1) ? xv : p1;
      p2 = (msk && q == 2) ? xv : p2;
      p3 = (msk && q == 3) ? xv : p3;
      xv = msk ? -INFINITY : xv;                     // exclude from negatives
      const float mn = fmaxf(m[g], xv);
      S[g] = S[g] * __expf(m[g] - mn) + __expf(xv - mn);
      m[g] = mn;
    }
  }

  // Merge (m,S) across the 16 column residues (stays within each half-wave).
#pragma unroll
  for (int off = 1; off < 16; off <<= 1) {
#pragma unroll
    for (int g = 0; g < 8; ++g) {
      const float m2 = __shfl_xor(m[g], off, 32);
      const float S2 = __shfl_xor(S[g], off, 32);
      const float mn = fmaxf(m[g], m2);
      S[g] = S[g] * __expf(m[g] - mn) + S2 * __expf(m2 - mn);
      m[g] = mn;
    }
  }

  // Designated lanes (g_diag valid: lanes 0-7 and 24-31) finish row M=l16.
  float acc = 0.0f;
  if (g_diag >= 0 && g_diag < 8) {
    const float m0v   = m[g_diag];
    const float S0v   = S[g_diag];
    const int   q_row = r0 >> 11;
    const float pv[4] = {p0, p1, p2, p3};
#pragma unroll
    for (int q = 0; q < N_T; ++q) {
      if (q != q_row) {
        const float p  = pv[q];
        const float mt = fmaxf(p, m0v);
        const float lse =
            mt + __logf(__expf(p - mt) + S0v * __expf(m0v - mt));
        acc += lse - p;
      }
    }
  }

  // Reduce loss contribution across the wave, one atomic per wave.
#pragma unroll
  for (int off = 1; off < 32; off <<= 1) acc += __shfl_xor(acc, off, 32);
  if (lane == 0) atomicAdd(out, acc * (1.0f / (float)KPOS));
}

// ---------------------------------------------------------------------------
extern "C" void kernel_launch(void* const* d_in, const int* in_sizes, int n_in,
                              void* d_out, int out_size, void* d_ws, size_t ws_size,
                              hipStream_t stream) {
  const float* z   = (const float*)d_in[0];        // (4, 2048, 128) f32
  _Float16*    snh = (_Float16*)d_ws;              // 8192 x 128 f16 (2 MB)
  float*       out = (float*)d_out;                // scalar loss

  ClBlockLoss_zero_kernel<<<1, 1, 0, stream>>>(out);
  ClBlockLoss_normalize_kernel<<<KTOT / 8, 256, 0, stream>>>(z, snh);
  ClBlockLoss_simloss_kernel<<<(KTOT / 16) / 8, 256, 0, stream>>>(snh, out);
}
